// GAT_1_24086176596168
// MI455X (gfx1250) — matrix operand order repaired and checked by reading, not compiled, not run on hardware
//
#include <hip/hip_runtime.h>
#include <hip/hip_bf16.h>

// ---------------------------------------------------------------------------
// Types for CDNA5 WMMA
// ---------------------------------------------------------------------------
typedef __attribute__((ext_vector_type(16))) __bf16 v16bf;
typedef __attribute__((ext_vector_type(8)))  float  v8f;

union BF16Frag {
    v16bf v;
    unsigned int w[8];        // packed pairs: w[i] = half(2i) | half(2i+1)<<16
    unsigned short u[16];
};

__device__ __forceinline__ unsigned short f32_to_bf16_rne(float f) {
    unsigned int u = __float_as_uint(f);
    u += 0x7FFFu + ((u >> 16) & 1u);     // round-to-nearest-even
    return (unsigned short)(u >> 16);
}

// pack two floats into one dword of bf16 pair (lo, hi)
__device__ __forceinline__ unsigned int pk_bf16(float lo, float hi) {
    return (unsigned int)f32_to_bf16_rne(lo) |
           ((unsigned int)f32_to_bf16_rne(hi) << 16);
}

__device__ __forceinline__ float leaky02(float x) {
    return x > 0.0f ? x : 0.2f * x;
}

// float atomic max via monotone int/uint encoding (init with -3e38)
__device__ __forceinline__ void atomicMaxF32(float* addr, float val) {
    if (val >= 0.0f) atomicMax((int*)addr, __float_as_int(val));
    else             atomicMin((unsigned int*)addr, __float_as_uint(val));
}

// ---------------------------------------------------------------------------
// WMMA GEMM:  C[M,N] = A[M,K] @ B[K,N]   (f32 in/out, bf16 WMMA math)
//   - B staged in LDS in FRAGMENT-MAJOR layout: for each (ktile, ntile) the
//     exact per-lane 16x16x32 B fragment is contiguous (32 B/lane, aligned),
//     so the inner-loop B load is one 32-byte LDS read (2x ds_load_b128).
//   - 8 waves / block; each wave owns one 16-row tile, NT column tiles.
//   - A fragment per 16x32 k-slice loaded once (global b128) and reused
//     across all NT column tiles.
// ---------------------------------------------------------------------------
template <int NT>
__global__ __launch_bounds__(256) void wmma_gemm_bf16_kernel(
    const float* __restrict__ A, const float* __restrict__ B,
    float* __restrict__ C, int M, int N, int K)
{
    __shared__ alignas(16) unsigned short sB[8192];   // 16 KB max (Kpad*N halves)

    const int Kpad   = (K + 31) & ~31;
    const int ktiles = Kpad >> 5;

    // ---- stage B into fragment-major layout: sB[((kt*NT+nt)*32 + lane)*16 + j]
    //      holds bf16( B[kt*32 + (lane>>4)*8 + (j&7) + (j>>3)*16, nt*16 + (lane&15)] )
    const int totalFrag = ktiles * NT * 512;          // 512 halves per fragment
    for (int i = threadIdx.x; i < totalFrag; i += blockDim.x) {
        int frag = i >> 9;                            // (kt*NT + nt)
        int rem  = i & 511;
        int ln   = rem >> 4;                          // lane 0..31
        int j    = rem & 15;
        int kt   = frag / NT;
        int nt   = frag - kt * NT;
        int k    = kt * 32 + (ln >> 4) * 8 + (j & 7) + (j >> 3) * 16;
        int n    = nt * 16 + (ln & 15);
        float v  = (k < K) ? B[(size_t)k * N + n] : 0.0f;
        sB[i] = f32_to_bf16_rne(v);
    }
    __syncthreads();

    const int lane   = threadIdx.x & 31;
    const int wave   = threadIdx.x >> 5;
    const int mtile  = blockIdx.x * 8 + wave;
    const int Mtiles = (M + 15) >> 4;
    if (mtile >= Mtiles) return;                      // wave-uniform exit

    const int row0 = mtile * 16;
    const int rlo  = lane & 15;                       // row within tile (A), col (C)
    const int kgrp = lane >> 4;                       // 0: K 0-7,16-23, 1: K 8-15,24-31

    v8f acc[NT];
#pragma unroll
    for (int nt = 0; nt < NT; ++nt) acc[nt] = (v8f){0,0,0,0,0,0,0,0};

    const float* __restrict__ arow = A + (size_t)(row0 + rlo) * K;
    const unsigned short* __restrict__ bfrag = sB + (lane << 4);   // this lane's slot

    for (int kt = 0; kt < ktiles; ++kt) {
        // ---- A fragment (16x32 bf16, ISA layout) ----
        BF16Frag a;
        const int kbase = kt * 32 + kgrp * 8;         // u[0..7]  <- K = kbase..+7
                                                      // u[8..15] <- K = kbase+16..+23
        if (kt * 32 + 32 <= K) {                      // fast path: full tile in range
            float4 f0 = *(const float4*)(arow + kbase);
            float4 f1 = *(const float4*)(arow + kbase + 4);
            float4 f2 = *(const float4*)(arow + kbase + 16);
            float4 f3 = *(const float4*)(arow + kbase + 20);
            a.w[0] = pk_bf16(f0.x, f0.y);  a.w[1] = pk_bf16(f0.z, f0.w);
            a.w[2] = pk_bf16(f1.x, f1.y);  a.w[3] = pk_bf16(f1.z, f1.w);
            a.w[4] = pk_bf16(f2.x, f2.y);  a.w[5] = pk_bf16(f2.z, f2.w);
            a.w[6] = pk_bf16(f3.x, f3.y);  a.w[7] = pk_bf16(f3.z, f3.w);
        } else {                                      // tail: zero-pad K
#pragma unroll
            for (int j = 0; j < 4; ++j) {
                int k0 = kbase + 2 * j, k1 = kbase + 16 + 2 * j;
                float lo0 = (k0     < K) ? arow[k0]     : 0.0f;
                float hi0 = (k0 + 1 < K) ? arow[k0 + 1] : 0.0f;
                float lo1 = (k1     < K) ? arow[k1]     : 0.0f;
                float hi1 = (k1 + 1 < K) ? arow[k1 + 1] : 0.0f;
                a.w[j]     = pk_bf16(lo0, hi0);
                a.w[4 + j] = pk_bf16(lo1, hi1);
            }
        }
        // ---- B fragments: single 32-byte LDS read per (kt, nt) + WMMA ----
#pragma unroll
        for (int nt = 0; nt < NT; ++nt) {
            v16bf b = *(const v16bf*)(bfrag + (((kt * NT + nt) << 9)));
            acc[nt] = __builtin_amdgcn_wmma_f32_16x16x32_bf16(
                false, a.v, false, b, (short)0, acc[nt], false, false);
        }
    }

    // ---- store C (lane 0-15: M=v ; lane 16-31: M=8+v) ----
    const int crow = row0 + kgrp * 8;
#pragma unroll
    for (int nt = 0; nt < NT; ++nt) {
        const int ncol = nt * 16 + rlo;
#pragma unroll
        for (int v = 0; v < 8; ++v)
            C[(size_t)(crow + v) * N + ncol] = acc[nt][v];
    }
}

// ---------------------------------------------------------------------------
// Per-(node,head) attention scores: s = <h[n,h,:], a[h,:]>
// ---------------------------------------------------------------------------
__global__ void score_kernel(const float* __restrict__ h,
                             const float* __restrict__ a_src,
                             const float* __restrict__ a_dst,
                             float* __restrict__ s_src, float* __restrict__ s_dst,
                             int Nn, int H, int C)
{
    int idx = blockIdx.x * blockDim.x + threadIdx.x;
    if (idx >= Nn * H) return;
    int n = idx / H, hh = idx - n * H;
    const float* row = h + (size_t)n * H * C + hh * C;
    const float* as  = a_src + hh * C;
    const float* ad  = a_dst + hh * C;
    float ss = 0.0f, sd = 0.0f;
    for (int c = 0; c < C; ++c) { float v = row[c]; ss += v * as[c]; sd += v * ad[c]; }
    s_src[idx] = ss;
    s_dst[idx] = sd;
}

__global__ void gat_init_kernel(float* __restrict__ m, float* __restrict__ denom,
                                float* __restrict__ acc, int NH, int NF)
{
    int i = blockIdx.x * blockDim.x + threadIdx.x;
    if (i < NH) { m[i] = -3.0e38f; denom[i] = 0.0f; }
    if (i < NF) acc[i] = 0.0f;
}

// pass 1: segment max of leaky-relu score over dst
__global__ void edge_max_kernel(const int* __restrict__ src, const int* __restrict__ dst,
                                const float* __restrict__ s_src, const float* __restrict__ s_dst,
                                float* __restrict__ m, int E, int Nn, int H)
{
    int idx = blockIdx.x * blockDim.x + threadIdx.x;
    int tot = (E + Nn) * H;
    if (idx >= tot) return;
    int e = idx / H, hh = idx - e * H;
    int s = (e < E) ? src[e] : (e - E);
    int d = (e < E) ? dst[e] : (e - E);
    float sc = leaky02(s_src[s * H + hh] + s_dst[d * H + hh]);
    atomicMaxF32(&m[d * H + hh], sc);
}

// pass 2: denom += e ; acc[dst] += e * h[src]   (unnormalized)
__global__ void edge_scatter_kernel(const int* __restrict__ src, const int* __restrict__ dst,
                                    const float* __restrict__ s_src, const float* __restrict__ s_dst,
                                    const float* __restrict__ m, const float* __restrict__ h,
                                    float* __restrict__ denom, float* __restrict__ acc,
                                    int E, int Nn, int H, int C)
{
    int idx = blockIdx.x * blockDim.x + threadIdx.x;
    int tot = (E + Nn) * H;
    if (idx >= tot) return;
    int e = idx / H, hh = idx - e * H;
    int s = (e < E) ? src[e] : (e - E);
    int d = (e < E) ? dst[e] : (e - E);
    float sc = leaky02(s_src[s * H + hh] + s_dst[d * H + hh]);
    float ex = __expf(sc - m[d * H + hh]);
    atomicAdd(&denom[d * H + hh], ex);
    const float* hrow = h   + (size_t)s * H * C + hh * C;
    float*       arow = acc + (size_t)d * H * C + hh * C;
    for (int c = 0; c < C; ++c)
        atomicAdd(&arow[c], ex * hrow[c]);
}

// pass 3: normalize + bias
__global__ void gat_finalize_kernel(float* __restrict__ acc, const float* __restrict__ denom,
                                    const float* __restrict__ bias, int Nn, int H, int C)
{
    int idx = blockIdx.x * blockDim.x + threadIdx.x;
    int HC = H * C;
    if (idx >= Nn * HC) return;
    int n = idx / HC, f = idx - n * HC;
    int hh = f / C;
    acc[idx] = acc[idx] / denom[n * H + hh] + bias[f];
}

// ---------------------------------------------------------------------------
// BatchNorm: one block per channel (column reduce), then fused apply+ReLU
// ---------------------------------------------------------------------------
__global__ __launch_bounds__(256) void bn_stats_kernel(const float* __restrict__ x,
                                                       float* __restrict__ mu, float* __restrict__ var,
                                                       int Nn, int F)
{
    int c = blockIdx.x;
    float s = 0.0f, s2 = 0.0f;
    for (int n = threadIdx.x; n < Nn; n += blockDim.x) {
        float v = x[(size_t)n * F + c];
        s += v; s2 += v * v;
    }
    __shared__ float sh[256], sh2[256];
    sh[threadIdx.x] = s; sh2[threadIdx.x] = s2;
    __syncthreads();
    for (int st = 128; st > 0; st >>= 1) {
        if (threadIdx.x < st) { sh[threadIdx.x] += sh[threadIdx.x + st];
                                sh2[threadIdx.x] += sh2[threadIdx.x + st]; }
        __syncthreads();
    }
    if (threadIdx.x == 0) {
        float mean = sh[0] / (float)Nn;
        mu[c]  = mean;
        var[c] = sh2[0] / (float)Nn - mean * mean;   // biased, like torch BN
    }
}

__global__ void bn_relu_kernel(float* __restrict__ x, const float* __restrict__ mu,
                               const float* __restrict__ var, const float* __restrict__ g,
                               const float* __restrict__ be, int Nn, int F)
{
    int idx = blockIdx.x * blockDim.x + threadIdx.x;
    if (idx >= Nn * F) return;
    int c = idx % F;
    float y = g[c] * (x[idx] - mu[c]) * __frsqrt_rn(var[c] + 1e-5f) + be[c];
    x[idx] = y > 0.0f ? y : 0.0f;
}

// small dense linear: y[N,Fout] = x[N,Fin] @ W[Fin,Fout] + b
__global__ void linear_kernel(const float* __restrict__ x, const float* __restrict__ W,
                              const float* __restrict__ b, float* __restrict__ y,
                              int Nn, int Fin, int Fout)
{
    int idx = blockIdx.x * blockDim.x + threadIdx.x;
    if (idx >= Nn * Fout) return;
    int n = idx / Fout, j = idx - n * Fout;
    const float* row = x + (size_t)n * Fin;
    float s = b[j];
    for (int k = 0; k < Fin; ++k) s += row[k] * W[k * Fout + j];
    y[idx] = s;
}

// ---------------------------------------------------------------------------
// Pooling + classifier head
// ---------------------------------------------------------------------------
__global__ void pool_init_kernel(float* __restrict__ sums, float* __restrict__ cnts,
                                 int G, int F)
{
    int i = blockIdx.x * blockDim.x + threadIdx.x;
    if (i < G * F) sums[i] = 0.0f;
    if (i < G)     cnts[i] = 0.0f;
}

__global__ void pool_accum_kernel(const float* __restrict__ h, const int* __restrict__ batch,
                                  float* __restrict__ sums, float* __restrict__ cnts,
                                  int Nn, int F)
{
    int idx = blockIdx.x * blockDim.x + threadIdx.x;
    if (idx >= Nn * F) return;
    int n = idx / F, c = idx - n * F;
    int g = batch[n];
    atomicAdd(&sums[(size_t)g * F + c], h[idx]);
    if (c == 0) atomicAdd(&cnts[g], 1.0f);
}

__global__ void head_kernel(const float* __restrict__ sums, const float* __restrict__ cnts,
                            const float* __restrict__ oW, const float* __restrict__ ob,
                            float* __restrict__ out, int G, int F, int Cls)
{
    int idx = blockIdx.x * blockDim.x + threadIdx.x;
    if (idx >= G * Cls) return;
    int g = idx / Cls, j = idx - g * Cls;
    float inv = 1.0f / fmaxf(cnts[g], 1.0f);
    const float* prow = sums + (size_t)g * F;
    float s = ob[j];
    for (int k = 0; k < F; ++k) s += (prow[k] * inv) * oW[k * Cls + j];
    out[idx] = s;
}

// ---------------------------------------------------------------------------
// Host orchestration
// ---------------------------------------------------------------------------
static inline int cdiv(int a, int b) { return (a + b - 1) / b; }

static void launch_gemm(hipStream_t st, const float* A, const float* B, float* C,
                        int M, int N, int K)
{
    int blocks = cdiv(cdiv(M, 16), 8);
    if (N == 96)
        wmma_gemm_bf16_kernel<6><<<blocks, 256, 0, st>>>(A, B, C, M, N, K);
    else
        wmma_gemm_bf16_kernel<4><<<blocks, 256, 0, st>>>(A, B, C, M, N, K);
}

static void run_gat_layer(hipStream_t st, const float* X, int Fin,
                          const float* W, const float* as, const float* ad, const float* bias,
                          int H, int C, const int* src, const int* dst, int E, int Nn,
                          float* P, float* ACC,
                          float* ssrc, float* sdst, float* mbuf, float* dbuf)
{
    const int HC = H * C;
    launch_gemm(st, X, W, P, Nn, HC, Fin);
    score_kernel<<<cdiv(Nn * H, 256), 256, 0, st>>>(P, as, ad, ssrc, sdst, Nn, H, C);
    gat_init_kernel<<<cdiv(Nn * HC, 256), 256, 0, st>>>(mbuf, dbuf, ACC, Nn * H, Nn * HC);
    int tot = (E + Nn) * H;
    edge_max_kernel<<<cdiv(tot, 256), 256, 0, st>>>(src, dst, ssrc, sdst, mbuf, E, Nn, H);
    edge_scatter_kernel<<<cdiv(tot, 256), 256, 0, st>>>(src, dst, ssrc, sdst, mbuf, P,
                                                        dbuf, ACC, E, Nn, H, C);
    gat_finalize_kernel<<<cdiv(Nn * HC, 256), 256, 0, st>>>(ACC, dbuf, bias, Nn, H, C);
}

static void run_bn_relu(hipStream_t st, float* X, const float* g, const float* be,
                        int Nn, int F, float* mu, float* var)
{
    bn_stats_kernel<<<F, 256, 0, st>>>(X, mu, var, Nn, F);
    bn_relu_kernel<<<cdiv(Nn * F, 256), 256, 0, st>>>(X, mu, var, g, be, Nn, F);
}

extern "C" void kernel_launch(void* const* d_in, const int* in_sizes, int n_in,
                              void* d_out, int out_size, void* d_ws, size_t ws_size,
                              hipStream_t stream)
{
    // ---- inputs (setup_inputs order) ----
    const float* x    = (const float*)d_in[0];
    const int*   ei   = (const int*)d_in[1];
    const int*   batch= (const int*)d_in[2];
    const float* W1   = (const float*)d_in[3];
    const float* a1s  = (const float*)d_in[4];
    const float* a1d  = (const float*)d_in[5];
    const float* b1   = (const float*)d_in[6];
    const float* g1   = (const float*)d_in[7];
    const float* be1  = (const float*)d_in[8];
    const float* lW1  = (const float*)d_in[9];
    const float* lb1  = (const float*)d_in[10];
    const float* gl1  = (const float*)d_in[11];
    const float* bel1 = (const float*)d_in[12];
    const float* W2   = (const float*)d_in[13];
    const float* a2s  = (const float*)d_in[14];
    const float* a2d  = (const float*)d_in[15];
    const float* b2   = (const float*)d_in[16];
    const float* g2   = (const float*)d_in[17];
    const float* be2  = (const float*)d_in[18];
    const float* lW2  = (const float*)d_in[19];
    const float* lb2  = (const float*)d_in[20];
    const float* gl2  = (const float*)d_in[21];
    const float* bel2 = (const float*)d_in[22];
    const float* W3   = (const float*)d_in[23];
    const float* a3s  = (const float*)d_in[24];
    const float* a3d  = (const float*)d_in[25];
    const float* b3   = (const float*)d_in[26];
    const float* g3   = (const float*)d_in[27];
    const float* be3  = (const float*)d_in[28];
    const float* oW   = (const float*)d_in[29];
    const float* ob   = (const float*)d_in[30];
    float* out = (float*)d_out;

    const int Nn = in_sizes[0] / 128;       // 100000
    const int E  = in_sizes[1] / 2;         // 1600000
    const int G  = 512, CLS = 10;
    const int* src = ei;
    const int* dst = ei + E;

    // ---- workspace carving (256B-aligned) ----
    char* w = (char*)d_ws;
    auto carve = [&](size_t bytes) -> float* {
        float* p = (float*)w;
        w += (bytes + 255) & ~(size_t)255;
        return p;
    };
    float* P    = carve((size_t)Nn * 96 * 4);   // projections
    float* ACC  = carve((size_t)Nn * 96 * 4);   // GAT output / features
    float* T    = carve((size_t)Nn * 96 * 4);   // post-linear features
    float* ssrc = carve((size_t)Nn * 8 * 4);
    float* sdst = carve((size_t)Nn * 8 * 4);
    float* mbuf = carve((size_t)Nn * 8 * 4);
    float* dbuf = carve((size_t)Nn * 8 * 4);
    float* mu   = carve(96 * 4);
    float* var  = carve(96 * 4);
    float* psum = carve((size_t)G * 96 * 4);
    float* pcnt = carve((size_t)G * 4);
    (void)ws_size; (void)n_in; (void)out_size;

    // ---- layer 1: GAT 128 -> 8x8, BN+ReLU, linear 64->8, BN+ReLU ----
    run_gat_layer(stream, x, 128, W1, a1s, a1d, b1, 8, 8, src, dst, E, Nn,
                  P, ACC, ssrc, sdst, mbuf, dbuf);
    run_bn_relu(stream, ACC, g1, be1, Nn, 64, mu, var);
    linear_kernel<<<cdiv(Nn * 8, 256), 256, 0, stream>>>(ACC, lW1, lb1, T, Nn, 64, 8);
    run_bn_relu(stream, T, gl1, bel1, Nn, 8, mu, var);

    // ---- layer 2: GAT 8 -> 4x16, BN+ReLU, linear 64->16, BN+ReLU ----
    run_gat_layer(stream, T, 8, W2, a2s, a2d, b2, 4, 16, src, dst, E, Nn,
                  P, ACC, ssrc, sdst, mbuf, dbuf);
    run_bn_relu(stream, ACC, g2, be2, Nn, 64, mu, var);
    linear_kernel<<<cdiv(Nn * 16, 256), 256, 0, stream>>>(ACC, lW2, lb2, T, Nn, 64, 16);
    run_bn_relu(stream, T, gl2, bel2, Nn, 16, mu, var);

    // ---- layer 3: GAT 16 -> 4x24, BN+ReLU ----
    run_gat_layer(stream, T, 16, W3, a3s, a3d, b3, 4, 24, src, dst, E, Nn,
                  P, ACC, ssrc, sdst, mbuf, dbuf);
    run_bn_relu(stream, ACC, g3, be3, Nn, 96, mu, var);

    // ---- mean pool over batch + classifier head ----
    pool_init_kernel<<<cdiv(G * 96, 256), 256, 0, stream>>>(psum, pcnt, G, 96);
    pool_accum_kernel<<<cdiv(Nn * 96, 256), 256, 0, stream>>>(ACC, batch, psum, pcnt, Nn, 96);
    head_kernel<<<cdiv(G * CLS, 256), 256, 0, stream>>>(psum, pcnt, oW, ob, out, G, 96, CLS);
}
